// CrossAttention_11922829214236
// MI455X (gfx1250) — matrix operand order repaired and checked by reading, not compiled, not run on hardware
//
#include <hip/hip_runtime.h>
#include <hip/hip_bf16.h>
#include <stdint.h>

#define HEADS   8
#define DHEAD   160
#define QDIM    1280
#define CDIM    768
#define INNER   1280
#define NQ      4096
#define NKV     77
#define BATCH   8
#define ATTN_SCALE 0.07905694150420949f   // 160^-0.5

typedef __attribute__((ext_vector_type(16))) __bf16 v16bf;
typedef __attribute__((ext_vector_type(8)))  float  v8f;
typedef int v4i __attribute__((vector_size(16)));   // matches builtin param type

#if __has_builtin(__builtin_amdgcn_global_load_async_to_lds_b128) && \
    __has_builtin(__builtin_amdgcn_s_wait_asynccnt)
#define USE_ASYNC 1
#else
#define USE_ASYNC 0
#endif

struct BF16x16 { uint4 lo, hi; };

__device__ __forceinline__ unsigned short f2bf(float f) {
  unsigned u = __builtin_bit_cast(unsigned, f);
  unsigned r = u + 0x7FFFu + ((u >> 16) & 1u);     // round-to-nearest-even
  return (unsigned short)(r >> 16);
}

__device__ __forceinline__ unsigned pk2(float a, float b) {
  return (unsigned)f2bf(a) | ((unsigned)f2bf(b) << 16);
}

// One 16-byte chunk global -> LDS. Async (ASYNCcnt-tracked, bypasses VGPRs)
// when the CDNA5 builtins are available, else synchronous copy.
__device__ __forceinline__ void copy16(const unsigned short* __restrict__ gsrc,
                                       unsigned short* ldst) {
#if USE_ASYNC
  __builtin_amdgcn_global_load_async_to_lds_b128(
      (__attribute__((address_space(1))) v4i*)gsrc,
      (__attribute__((address_space(3))) v4i*)ldst, 0, 0);
#else
  *(uint4*)ldst = *(const uint4*)gsrc;
#endif
}

__device__ __forceinline__ void wait_async_le_4() {
#if USE_ASYNC
  __builtin_amdgcn_s_wait_asynccnt(4);
#endif
}
__device__ __forceinline__ void wait_async_0() {
#if USE_ASYNC
  __builtin_amdgcn_s_wait_asynccnt(0);
#endif
}

// Build a 16-element bf16 fragment from two 16-byte chunks (p1 = p0+16 for A
// and B fragments, element offsets).
__device__ __forceinline__ v16bf ldfrag(const unsigned short* p0, const unsigned short* p1) {
  BF16x16 s;
  s.lo = *(const uint4*)p0;
  s.hi = *(const uint4*)p1;
  return __builtin_bit_cast(v16bf, s);
}

__device__ __forceinline__ v8f wmma_bf16(v16bf a, v16bf b, v8f c) {
  return __builtin_amdgcn_wmma_f32_16x16x32_bf16(false, a, false, b, (short)0, c,
                                                 false, false);
}

// ---------------------------------------------------------------------------
// Prep: elementwise f32 -> bf16 (8 elements / thread)
// ---------------------------------------------------------------------------
__global__ __launch_bounds__(256) void convert_f32_bf16(
    const float* __restrict__ src, unsigned short* __restrict__ dst, int ngroups)
{
  const int i = blockIdx.x * 256 + threadIdx.x;
  if (i < ngroups) {
    const float4* s = (const float4*)src + (size_t)i * 2;
    float4 a = s[0], b = s[1];
    uint4 r;
    r.x = pk2(a.x, a.y); r.y = pk2(a.z, a.w);
    r.z = pk2(b.x, b.y); r.w = pk2(b.z, b.w);
    ((uint4*)dst)[i] = r;
  }
}

// ---------------------------------------------------------------------------
// Prep: W f32 [K][N] -> WT bf16 [N][K] (tiled 32x32; K,N multiples of 32)
// ---------------------------------------------------------------------------
__global__ __launch_bounds__(256) void transpose_f32_bf16(
    const float* __restrict__ W, unsigned short* __restrict__ WT, int K, int N)
{
  __shared__ unsigned short t[32][33];
  const int n0 = blockIdx.x * 32, k0 = blockIdx.y * 32;
  const int tx = threadIdx.x, ty = threadIdx.y;   // 32 x 8
#pragma unroll
  for (int i = 0; i < 4; ++i)
    t[ty + i * 8][tx] = f2bf(W[(size_t)(k0 + ty + i * 8) * N + n0 + tx]);
  __syncthreads();
#pragma unroll
  for (int i = 0; i < 4; ++i)
    WT[(size_t)(n0 + ty + i * 8) * K + k0 + tx] = t[tx][ty + i * 8];
}

// ---------------------------------------------------------------------------
// GEMM: Out[M,1280] = A[M,K](bf16) @ BT[1280,K](bf16, N-major)
// Block tile 128x128, BK=32, 256 thr = 8 waves (4M x 2N), 32x64 per wave.
// Double-buffered LDS; async global->LDS staging pipelined one K-step ahead
// (uniform 4 async ops / wave / step so `s_wait_asynccnt 4` retires the
// previous step's tile).
//   HEAD_SPLIT=true : store bf16 to [B, H, rowsPerB, 160]
//   HEAD_SPLIT=false: store f32 [M,1280] + bias
// ---------------------------------------------------------------------------
template <bool HEAD_SPLIT>
__global__ __launch_bounds__(256) void gemm_bf16_wmma(
    const unsigned short* __restrict__ A, const unsigned short* __restrict__ BT,
    const float* __restrict__ bias, void* __restrict__ Outp,
    int M, int K, int rowsPerB)
{
  __shared__ __align__(16) unsigned short lds_a[2][128 * 32];  // [row][k]
  __shared__ __align__(16) unsigned short lds_b[2][128 * 32];  // [n][k]

  const int tid  = threadIdx.x;
  const int lane = tid & 31;
  const int wave = tid >> 5;
  const int wm   = wave & 3;
  const int wn   = wave >> 2;
  const int hi   = lane >> 4;
  const int ln   = lane & 15;
  const int mbase = blockIdx.y * 128;
  const int nbase = blockIdx.x * 128;

  v8f acc[2][4];
#pragma unroll
  for (int i = 0; i < 2; ++i)
#pragma unroll
    for (int j = 0; j < 4; ++j) acc[i][j] = {};

  // Stage one 128x32 A tile + 128x32 BT tile: 4 b128 copies per thread.
  auto stage = [&](int kb, int buf) {
#pragma unroll
    for (int i = 0; i < 2; ++i) {
      const int c   = tid * 2 + i;          // 0..511
      const int row = c >> 2;
      const int sub = (c & 3) << 3;         // 0,8,16,24 elements (16B chunks)
      int gr = mbase + row;
      gr = (gr < M) ? gr : 0;               // clamp: keeps async count uniform;
                                            // OOB rows never stored (gm<M guard)
      copy16(A + (size_t)gr * K + kb * 32 + sub, &lds_a[buf][row * 32 + sub]);
      const int nr = nbase + row;           // N=1280 divisible by 128: in range
      copy16(BT + (size_t)nr * K + kb * 32 + sub, &lds_b[buf][row * 32 + sub]);
    }
  };

  const int ksteps = K >> 5;
  stage(0, 0);
  for (int kb = 0; kb < ksteps; ++kb) {
    const int cur = kb & 1;
    if (kb + 1 < ksteps) { stage(kb + 1, cur ^ 1); wait_async_le_4(); }
    else                 { wait_async_0(); }
    __syncthreads();

    v16bf afr[2], bfr[4];
#pragma unroll
    for (int ms = 0; ms < 2; ++ms) {
      const int row = wm * 32 + ms * 16 + ln;
      const unsigned short* p = &lds_a[cur][row * 32 + hi * 8];
      afr[ms] = ldfrag(p, p + 16);
    }
#pragma unroll
    for (int ns = 0; ns < 4; ++ns) {
      const int col = wn * 64 + ns * 16 + ln;
      const unsigned short* p = &lds_b[cur][col * 32 + hi * 8];
      bfr[ns] = ldfrag(p, p + 16);
    }
#pragma unroll
    for (int ms = 0; ms < 2; ++ms)
#pragma unroll
      for (int ns = 0; ns < 4; ++ns)
        acc[ms][ns] = wmma_bf16(afr[ms], bfr[ns], acc[ms][ns]);

    __syncthreads();
  }

#pragma unroll
  for (int ms = 0; ms < 2; ++ms) {
#pragma unroll
    for (int ns = 0; ns < 4; ++ns) {
      const int gmBase = mbase + wm * 32 + ms * 16 + hi * 8;
      const int gn     = nbase + wn * 64 + ns * 16 + ln;
      const int h = gn / DHEAD, d = gn - h * DHEAD;
#pragma unroll
      for (int j = 0; j < 8; ++j) {
        const int gm = gmBase + j;
        if (gm < M) {
          const float val = acc[ms][ns][j];
          if constexpr (HEAD_SPLIT) {
            const int b = gm / rowsPerB, r = gm - b * rowsPerB;
            ((unsigned short*)Outp)[((size_t)((b * HEADS + h) * rowsPerB) + r) *
                                        DHEAD + d] = f2bf(val);
          } else {
            ((float*)Outp)[(size_t)gm * 1280 + gn] = val + bias[gn];
          }
        }
      }
    }
  }
}

// ---------------------------------------------------------------------------
// Fused attention: block = one (b,h), 64 q rows (4 waves x 16-row tiles).
// K and V^T staged in LDS once; masked softmax over Nk=77 (padded 80/96).
// ---------------------------------------------------------------------------
__global__ __launch_bounds__(128) void attn_kernel(
    const unsigned short* __restrict__ q, const unsigned short* __restrict__ kk,
    const unsigned short* __restrict__ vv, unsigned short* __restrict__ O)
{
  __shared__ __align__(16) unsigned short k_lds[80 * 160];      // [krow][d]
  __shared__ __align__(16) unsigned short vT_lds[160 * 80];     // [d][krow]
  __shared__ __align__(16) unsigned short attn_lds[4][16 * 96]; // per-wave

  const int tid  = threadIdx.x;
  const int lane = tid & 31;
  const int wave = tid >> 5;
  const int hi   = lane >> 4;
  const int ln   = lane & 15;
  const int bh   = blockIdx.y;
  const int b    = bh >> 3, h = bh & 7;

  // K rows 0..76 are contiguous bf16 -> async b128 copy; pad rows 77..79 = 0.
  const unsigned short* kbase = kk + (size_t)bh * NKV * DHEAD;
  for (int c = tid; c < (NKV * DHEAD) / 8; c += 128)    // 1540 16B chunks
    copy16(kbase + c * 8, &k_lds[c * 8]);
  for (int idx = tid; idx < 3 * DHEAD; idx += 128)
    k_lds[NKV * DHEAD + idx] = 0;
  // V^T needs a transpose -> regular element copies.
  const unsigned short* vbase = vv + (size_t)bh * NKV * DHEAD;
  for (int idx = tid; idx < 160 * 80; idx += 128) {
    const int d = idx / 80, j = idx - d * 80;
    vT_lds[idx] = (j < NKV) ? vbase[j * DHEAD + d] : (unsigned short)0;
  }
  wait_async_0();
  __syncthreads();

  const int qrow0 = blockIdx.x * 64 + wave * 16;
  const unsigned short* qb = q + ((size_t)bh * NQ + qrow0) * DHEAD;

  v16bf qa[5];
#pragma unroll
  for (int c = 0; c < 5; ++c) {
    const unsigned short* p = qb + (size_t)ln * DHEAD + c * 32 + hi * 8;
    qa[c] = ldfrag(p, p + 16);
  }

  // sim = q @ K^T : 5 N-tiles x 5 K-chunks
  v8f sim[5];
#pragma unroll
  for (int nt = 0; nt < 5; ++nt) {
    v8f a = {};
#pragma unroll
    for (int c = 0; c < 5; ++c) {
      const unsigned short* p = &k_lds[(nt * 16 + ln) * 160 + c * 32 + hi * 8];
      a = wmma_bf16(qa[c], ldfrag(p, p + 16), a);
    }
    sim[nt] = a;
  }

  // Masked softmax; bf16 weights to per-wave LDS scratch [16][96].
  unsigned short* arow = attn_lds[wave];
#pragma unroll
  for (int j = 0; j < 8; ++j) {
    float vals[5];
    float m = -3.0e38f;
#pragma unroll
    for (int t = 0; t < 5; ++t) {
      const int col = t * 16 + ln;
      const float v = sim[t][j] * ATTN_SCALE;
      vals[t] = v;
      if (col < NKV) m = fmaxf(m, v);
    }
#pragma unroll
    for (int off = 8; off >= 1; off >>= 1) m = fmaxf(m, __shfl_xor(m, off, 32));
    float s = 0.f;
#pragma unroll
    for (int t = 0; t < 5; ++t) {
      const int col = t * 16 + ln;
      const float e = (col < NKV) ? __expf(vals[t] - m) : 0.f;
      vals[t] = e;
      s += e;
    }
#pragma unroll
    for (int off = 8; off >= 1; off >>= 1) s += __shfl_xor(s, off, 32);
    const float inv = 1.0f / s;
    const int row = hi * 8 + j;
#pragma unroll
    for (int t = 0; t < 5; ++t)
      arow[row * 96 + t * 16 + ln] = f2bf(vals[t] * inv);
    arow[row * 96 + 80 + ln] = 0;   // zero cols 80..95
  }

  // out = attn @ V : 10 N-tiles (d) x 3 K-chunks (k rows padded to 96).
  v16bf af[3];
#pragma unroll
  for (int c = 0; c < 3; ++c) {
    const unsigned short* p = &arow[ln * 96 + c * 32 + hi * 8];
    af[c] = ldfrag(p, p + 16);
  }
  const size_t orow = (size_t)(b * NQ + qrow0 + hi * 8);
#pragma unroll
  for (int nt = 0; nt < 10; ++nt) {
    v8f a = {};
#pragma unroll
    for (int c = 0; c < 3; ++c) {
      const unsigned short* p = &vT_lds[(nt * 16 + ln) * 80 + c * 32 + hi * 8];
      a = wmma_bf16(af[c], ldfrag(p, p + 16), a);
    }
    const int gcol = h * DHEAD + nt * 16 + ln;
#pragma unroll
    for (int j = 0; j < 8; ++j)
      O[(orow + j) * INNER + gcol] = f2bf(a[j]);
  }
}

// ---------------------------------------------------------------------------
extern "C" void kernel_launch(void* const* d_in, const int* in_sizes, int n_in,
                              void* d_out, int out_size, void* d_ws, size_t ws_size,
                              hipStream_t stream) {
  (void)in_sizes; (void)n_in; (void)out_size; (void)ws_size;
  const float* x   = (const float*)d_in[0];
  const float* ctx = (const float*)d_in[1];
  const float* Wq  = (const float*)d_in[2];
  const float* Wk  = (const float*)d_in[3];
  const float* Wv  = (const float*)d_in[4];
  const float* Wo  = (const float*)d_in[5];
  const float* bo  = (const float*)d_in[6];
  float* out = (float*)d_out;

  char* ws = (char*)d_ws;
  auto take = [&](size_t bytes) { char* p = ws; ws += (bytes + 255) & ~(size_t)255; return p; };
  unsigned short* xbf  = (unsigned short*)take((size_t)BATCH * NQ * QDIM * 2);
  unsigned short* cbf  = (unsigned short*)take((size_t)BATCH * NKV * CDIM * 2);
  unsigned short* WqT  = (unsigned short*)take((size_t)INNER * QDIM * 2);
  unsigned short* WkT  = (unsigned short*)take((size_t)INNER * CDIM * 2);
  unsigned short* WvT  = (unsigned short*)take((size_t)INNER * CDIM * 2);
  unsigned short* WoT  = (unsigned short*)take((size_t)QDIM * INNER * 2);
  unsigned short* q_ws = (unsigned short*)take((size_t)BATCH * NQ * INNER * 2);
  unsigned short* k_ws = (unsigned short*)take((size_t)BATCH * HEADS * NKV * DHEAD * 2);
  unsigned short* v_ws = (unsigned short*)take((size_t)BATCH * HEADS * NKV * DHEAD * 2);
  unsigned short* O_ws = (unsigned short*)take((size_t)BATCH * NQ * INNER * 2);

  // --- prep: bf16 conversions + weight transposes ---
  {
    const int gx = (BATCH * NQ * QDIM) / 8;
    convert_f32_bf16<<<dim3((gx + 255) / 256), dim3(256), 0, stream>>>(x, xbf, gx);
    const int gc = (BATCH * NKV * CDIM) / 8;
    convert_f32_bf16<<<dim3((gc + 255) / 256), dim3(256), 0, stream>>>(ctx, cbf, gc);
    const dim3 tb(32, 8);
    transpose_f32_bf16<<<dim3(INNER / 32, QDIM / 32), tb, 0, stream>>>(Wq, WqT, QDIM, INNER);
    transpose_f32_bf16<<<dim3(INNER / 32, CDIM / 32), tb, 0, stream>>>(Wk, WkT, CDIM, INNER);
    transpose_f32_bf16<<<dim3(INNER / 32, CDIM / 32), tb, 0, stream>>>(Wv, WvT, CDIM, INNER);
    transpose_f32_bf16<<<dim3(QDIM / 32, INNER / 32), tb, 0, stream>>>(Wo, WoT, INNER, QDIM);
  }

  const dim3 blk(256);
  // q = x @ Wq   -> head-split bf16 [B,H,4096,160]
  gemm_bf16_wmma<true><<<dim3(10, 256), blk, 0, stream>>>(
      xbf, WqT, nullptr, q_ws, BATCH * NQ, QDIM, NQ);
  // k = ctx @ Wk -> head-split bf16 [B,H,77,160]
  gemm_bf16_wmma<true><<<dim3(10, 5), blk, 0, stream>>>(
      cbf, WkT, nullptr, k_ws, BATCH * NKV, CDIM, NKV);
  // v = ctx @ Wv
  gemm_bf16_wmma<true><<<dim3(10, 5), blk, 0, stream>>>(
      cbf, WvT, nullptr, v_ws, BATCH * NKV, CDIM, NKV);
  // fused attention -> O bf16 [B, Nq, 1280]
  attn_kernel<<<dim3(NQ / 64, BATCH * HEADS), dim3(128), 0, stream>>>(
      q_ws, k_ws, v_ws, O_ws);
  // out = O @ Wo + bo -> f32
  gemm_bf16_wmma<false><<<dim3(10, 256), blk, 0, stream>>>(
      O_ws, WoT, bo, out, BATCH * NQ, INNER, 1);
}
